// LinearCrossAttention_23132693856724
// MI455X (gfx1250) — matrix operand order repaired and checked by reading, not compiled, not run on hardware
//
#include <hip/hip_runtime.h>
#include <hip/hip_bf16.h>

// B=1024, S=64, E=1024, H=16, D=64. M = B*S = 65536.
// Stage 0: transpose+convert weights W[K,N]f32 -> Wt[N,K]bf16 (async-copy friendly).
// Stage 1: Q/K/V projections (bf16 WMMA GEMM, ReLU on Q,K) -> bf16 [M,E].
// Stage 2: per-head 64x64 causal attention + softmax (WMMA) -> bf16 [M,E].
// Stage 3: output projection GEMM (fully async A+B) -> fp32 d_out.

typedef __bf16 bf16_t;
typedef __attribute__((ext_vector_type(16))) __bf16 v16bf;
typedef __attribute__((ext_vector_type(8)))  __bf16 v8bf;
typedef __attribute__((ext_vector_type(4)))  __bf16 v4bf;
typedef __attribute__((ext_vector_type(8)))  float  v8f;

#define BM 128
#define BN 128
#define BK 64
#define LDK 72       // 64 + 8 bf16 pad -> 144B row stride (16B aligned rows)
#define LDQ 72

// ---- CDNA5 async global->LDS copy (ASYNCcnt) ----
__device__ __forceinline__ uint32_t lds_off(const void* p) {
  // generic pointers to LDS carry the LDS byte offset in the low 32 bits
  return (uint32_t)(uintptr_t)p;
}
__device__ __forceinline__ void async_copy_b128(uint32_t ldsoff, const void* gaddr) {
  asm volatile("global_load_async_to_lds_b128 %0, %1, off"
               :: "v"(ldsoff), "v"(gaddr) : "memory");
}
template <int N>
__device__ __forceinline__ void wait_asynccnt() {
  asm volatile("s_wait_asynccnt %0" :: "n"(N) : "memory");
}

// ---- WMMA fragment loaders (ISA 7.12.2 layouts, wave32) ----
// A: 16x32 bf16. lane%16 = M row; lanes 0-15 hold K 0..7 & 16..23,
// lanes 16-31 hold K 8..15 & 24..31. base points at (row0,k0); ld in elems.
__device__ __forceinline__ v16bf load_frag_a(const bf16_t* base, int ld, int lane) {
  const int r  = lane & 15;
  const int kb = (lane >> 4) << 3;
  const bf16_t* p = base + r * ld + kb;
  v16bf a;
#pragma unroll
  for (int i = 0; i < 8; ++i) a[i] = p[i];
#pragma unroll
  for (int i = 0; i < 8; ++i) a[8 + i] = p[16 + i];
  return a;
}
// B: 32x16 bf16 stored in LDS as [n][k]. lane%16 = N col; lanes 0-15 hold
// K 0..7 & 16..23, lanes 16-31 hold K 8..15 & 24..31. bt points at (n0,k0).
__device__ __forceinline__ v16bf load_frag_b(const bf16_t* bt, int ld, int lane) {
  const int n  = lane & 15;
  const int kb = (lane >> 4) << 3;
  const bf16_t* p = bt + n * ld + kb;
  v16bf b;
#pragma unroll
  for (int i = 0; i < 8; ++i) b[i] = p[i];
#pragma unroll
  for (int i = 0; i < 8; ++i) b[8 + i] = p[16 + i];
  return b;
}

// ---- Stage 0: W[K,N] f32 -> Wt[N,K] bf16 (32x32 tiles via LDS) ----
__global__ __launch_bounds__(256) void wconv_kernel(
    const float* __restrict__ W, bf16_t* __restrict__ Wt, int K, int N) {
  __shared__ float Ts[32][33];
  const int t  = threadIdx.x;
  const int n0 = blockIdx.x * 32, k0 = blockIdx.y * 32;
  {
    const int kr = t >> 3, c4 = (t & 7) * 4;
    const float4 v = *(const float4*)(W + (size_t)(k0 + kr) * N + n0 + c4);
    Ts[kr][c4 + 0] = v.x; Ts[kr][c4 + 1] = v.y;
    Ts[kr][c4 + 2] = v.z; Ts[kr][c4 + 3] = v.w;
  }
  __syncthreads();
  {
    const int nr = t >> 3, c4 = (t & 7) * 4;
    v4bf o;
#pragma unroll
    for (int j = 0; j < 4; ++j) o[j] = (bf16_t)Ts[c4 + j][nr];
    *(v4bf*)(Wt + (size_t)(n0 + nr) * K + k0 + c4) = o;
  }
}

// ---- Stage 1 / Stage 3: double-buffered bf16 WMMA GEMM ----
// C = A@W + bias.  Wt: [N,K] bf16 (async-copied).
// ASYNC_A=1: A is bf16 [M,K], async-copied. ASYNC_A=0: A is f32, VGPR-staged.
// OUT_F32=0: bf16 [M,N] out (+optional ReLU). OUT_F32=1: f32 [M,N] out.
template <int ASYNC_A, int RELU, int OUT_F32>
__global__ __launch_bounds__(256) void gemm_wmma2(
    const void* __restrict__ Ap, const bf16_t* __restrict__ Wt,
    const float* __restrict__ bias, bf16_t* __restrict__ out_bf,
    float* __restrict__ out_f, int M, int N, int K) {
  __shared__ __align__(16) bf16_t As[2][BM * LDK];
  __shared__ __align__(16) bf16_t Bs[2][BN * LDK];

  const int tid  = threadIdx.x;
  const int lane = tid & 31;
  const int w    = tid >> 5;
  const int wm   = w >> 1;   // 0..3 -> 32-row slab
  const int wn   = w & 1;    // 0..1 -> 64-col slab
  const int n0   = blockIdx.x * BN;
  const int m0   = blockIdx.y * BM;
  const int nIter = K / BK;
  // 128x64 tile = 1024 chunks of 8 bf16 (16B); 4 chunks/thread.
  int crow[4], ccol[4];
#pragma unroll
  for (int u = 0; u < 4; ++u) {
    const int c = tid * 4 + u;
    crow[u] = c >> 3;            // 8 chunks per 64-elem row
    ccol[u] = (c & 7) * 8;
  }
  constexpr int PEND = ASYNC_A ? 8 : 4;  // async issues per thread per buffer

  const bf16_t* Ab16 = (const bf16_t*)Ap;
  const float*  Af32 = (const float*)Ap;
  float4 s0[4], s1[4];  // fp32 A staging (ASYNC_A==0)

  v8f acc[2][4];
#pragma unroll
  for (int tm = 0; tm < 2; ++tm)
#pragma unroll
    for (int tn = 0; tn < 4; ++tn) acc[tm][tn] = (v8f){};

  auto issueB = [&](int buf, int kk) {
#pragma unroll
    for (int u = 0; u < 4; ++u)
      async_copy_b128(lds_off(&Bs[buf][crow[u] * LDK + ccol[u]]),
                      Wt + (size_t)(n0 + crow[u]) * K + kk + ccol[u]);
  };
  auto issueA = [&](int buf, int kk) {
#pragma unroll
    for (int u = 0; u < 4; ++u)
      async_copy_b128(lds_off(&As[buf][crow[u] * LDK + ccol[u]]),
                      Ab16 + (size_t)(m0 + crow[u]) * K + kk + ccol[u]);
  };
  auto loadA_f32 = [&](int kk) {
#pragma unroll
    for (int u = 0; u < 4; ++u) {
      const float* p = Af32 + (size_t)(m0 + crow[u]) * K + kk + ccol[u];
      s0[u] = *(const float4*)p;
      s1[u] = *(const float4*)(p + 4);
    }
  };
  auto storeA_f32 = [&](int buf) {
#pragma unroll
    for (int u = 0; u < 4; ++u) {
      v8bf o;
      o[0] = (bf16_t)s0[u].x; o[1] = (bf16_t)s0[u].y;
      o[2] = (bf16_t)s0[u].z; o[3] = (bf16_t)s0[u].w;
      o[4] = (bf16_t)s1[u].x; o[5] = (bf16_t)s1[u].y;
      o[6] = (bf16_t)s1[u].z; o[7] = (bf16_t)s1[u].w;
      *(v8bf*)(&As[buf][crow[u] * LDK + ccol[u]]) = o;
    }
  };
  auto compute = [&](int buf) {
    const bf16_t* Abase = &As[buf][(wm * 32) * LDK];
    const bf16_t* Bbase = &Bs[buf][(wn * 64) * LDK];
#pragma unroll
    for (int ks = 0; ks < 2; ++ks) {
      v16bf af[2], bf[4];
#pragma unroll
      for (int tm = 0; tm < 2; ++tm)
        af[tm] = load_frag_a(Abase + tm * 16 * LDK + ks * 32, LDK, lane);
#pragma unroll
      for (int tn = 0; tn < 4; ++tn)
        bf[tn] = load_frag_b(Bbase + tn * 16 * LDK + ks * 32, LDK, lane);
#pragma unroll
      for (int tm = 0; tm < 2; ++tm)
#pragma unroll
        for (int tn = 0; tn < 4; ++tn)
          acc[tm][tn] = __builtin_amdgcn_wmma_f32_16x16x32_bf16(
              false, af[tm], false, bf[tn], (short)0, acc[tm][tn], false, false);
    }
  };

  // prologue
  issueB(0, 0);
  if (ASYNC_A) issueA(0, 0); else loadA_f32(0);

  for (int i = 0; i < nIter; ++i) {
    const int buf = i & 1;
    const bool more = (i + 1) < nIter;
    if (!ASYNC_A) storeA_f32(buf);
    if (more) {
      issueB(buf ^ 1, (i + 1) * BK);
      if (ASYNC_A) issueA(buf ^ 1, (i + 1) * BK); else loadA_f32((i + 1) * BK);
    }
    if (more) wait_asynccnt<PEND>(); else wait_asynccnt<0>();
    __syncthreads();
    compute(buf);
    __syncthreads();
  }

  // epilogue: bias (+ReLU), store
  const int coff = lane & 15;
  const int roff = (lane >> 4) << 3;
#pragma unroll
  for (int tm = 0; tm < 2; ++tm) {
#pragma unroll
    for (int tn = 0; tn < 4; ++tn) {
      const int n = n0 + wn * 64 + tn * 16 + coff;
      const float bv = bias[n];
#pragma unroll
      for (int r = 0; r < 8; ++r) {
        const int m = m0 + wm * 32 + tm * 16 + roff + r;
        float v = acc[tm][tn][r] + bv;
        if (RELU) v = fmaxf(v, 0.0f);
        if (OUT_F32) out_f[(size_t)m * N + n] = v;
        else         out_bf[(size_t)m * N + n] = (bf16_t)v;
      }
    }
  }
}

// ---- Stage 2: per-head causal attention, 64x64, softmax, P@V ----
__global__ __launch_bounds__(256) void attn_head_kernel(
    const bf16_t* __restrict__ qm, const bf16_t* __restrict__ km,
    const bf16_t* __restrict__ vm, bf16_t* __restrict__ attn_out) {
  __shared__ __align__(16) bf16_t Qs[64 * LDQ];  // [q][d]; reused as P[q][kv]
  __shared__ __align__(16) bf16_t Ks[64 * LDQ];  // [kv][d]
  __shared__ __align__(16) bf16_t Vt[64 * LDQ];  // [d][kv] (transposed)
  __shared__ float Sf[64 * 65];

  const int tid   = threadIdx.x;
  const int lane  = tid & 31;
  const int w     = tid >> 5;
  const int batch = blockIdx.x >> 4;
  const int h     = blockIdx.x & 15;
  const size_t rowbase = (size_t)batch * 64 * 1024 + h * 64;  // into [M,E]

  // cooperative loads: 64 rows x 64 bf16 (head tile from [M,E]); V transposed
  const int c0 = tid * 2;
#pragma unroll
  for (int u = 0; u < 2; ++u) {
    const int c   = c0 + u;      // 512 chunks of 8
    const int row = c >> 3;      // 8 chunks per 64-elem row
    const int c8  = c & 7;
    const size_t g = rowbase + (size_t)row * 1024 + c8 * 8;
    *(v8bf*)&Qs[row * LDQ + c8 * 8] = *(const v8bf*)(qm + g);
    *(v8bf*)&Ks[row * LDQ + c8 * 8] = *(const v8bf*)(km + g);
    const v8bf vv = *(const v8bf*)(vm + g);
#pragma unroll
    for (int j = 0; j < 8; ++j) Vt[(c8 * 8 + j) * LDQ + row] = vv[j];
  }
  __syncthreads();

  // scores = Q @ K^T * scale, causal mask -> Sf
#pragma unroll
  for (int t = 2 * w; t < 2 * w + 2; ++t) {
    const int ti = t >> 2, tj = t & 3;
    v8f c = (v8f){};
#pragma unroll
    for (int ks = 0; ks < 2; ++ks) {
      const v16bf a = load_frag_a(Qs + ti * 16 * LDQ + ks * 32, LDQ, lane);
      const v16bf b = load_frag_b(Ks + tj * 16 * LDQ + ks * 32, LDQ, lane);
      c = __builtin_amdgcn_wmma_f32_16x16x32_bf16(false, a, false, b, (short)0, c, false, false);
    }
    const int kv   = tj * 16 + (lane & 15);
    const int roff = (lane >> 4) << 3;
#pragma unroll
    for (int r = 0; r < 8; ++r) {
      const int q = ti * 16 + roff + r;
      Sf[q * 65 + kv] = (kv <= q) ? c[r] * 0.125f : -3.0e38f;
    }
  }
  __syncthreads();

  // row softmax -> P (reusing Qs)
  if (tid < 64) {
    const int q = tid;
    float mx = -3.4e38f;
    for (int k2 = 0; k2 < 64; ++k2) mx = fmaxf(mx, Sf[q * 65 + k2]);
    float sum = 0.0f;
    for (int k2 = 0; k2 < 64; ++k2) sum += __expf(Sf[q * 65 + k2] - mx);
    const float inv = 1.0f / sum;
    for (int k2 = 0; k2 < 64; ++k2)
      Qs[q * LDQ + k2] = (bf16_t)(__expf(Sf[q * 65 + k2] - mx) * inv);
  }
  __syncthreads();

  // out = P @ V -> attn_out[M,E]
#pragma unroll
  for (int t = 2 * w; t < 2 * w + 2; ++t) {
    const int ti = t >> 2, tj = t & 3;
    v8f c = (v8f){};
#pragma unroll
    for (int ks = 0; ks < 2; ++ks) {
      const v16bf a = load_frag_a(Qs + ti * 16 * LDQ + ks * 32, LDQ, lane);
      const v16bf b = load_frag_b(Vt + tj * 16 * LDQ + ks * 32, LDQ, lane);
      c = __builtin_amdgcn_wmma_f32_16x16x32_bf16(false, a, false, b, (short)0, c, false, false);
    }
    const int d    = tj * 16 + (lane & 15);
    const int roff = (lane >> 4) << 3;
#pragma unroll
    for (int r = 0; r < 8; ++r) {
      const int q = ti * 16 + roff + r;
      attn_out[rowbase + (size_t)(ti * 16 + roff + r) * 1024 + (tj * 16 + (lane & 15))]
          = (bf16_t)c[r];
      (void)q; (void)d;
    }
  }
}

extern "C" void kernel_launch(void* const* d_in, const int* in_sizes, int n_in,
                              void* d_out, int out_size, void* d_ws, size_t ws_size,
                              hipStream_t stream) {
  (void)in_sizes; (void)n_in; (void)out_size; (void)ws_size;
  const float* query = (const float*)d_in[0];
  const float* key   = (const float*)d_in[1];
  const float* value = (const float*)d_in[2];
  const float* Wq = (const float*)d_in[3];  const float* bq = (const float*)d_in[4];
  const float* Wk = (const float*)d_in[5];  const float* bk = (const float*)d_in[6];
  const float* Wv = (const float*)d_in[7];  const float* bv = (const float*)d_in[8];
  const float* Wo = (const float*)d_in[9];  const float* bo = (const float*)d_in[10];

  const int M = 65536, N = 1024, K = 1024;
  char* ws = (char*)d_ws;
  const size_t seg  = (size_t)M * N * sizeof(bf16_t);  // 128 MiB
  const size_t wseg = (size_t)K * N * sizeof(bf16_t);  // 2 MiB
  bf16_t* qbf = (bf16_t*)(ws);
  bf16_t* kbf = (bf16_t*)(ws + seg);
  bf16_t* vbf = (bf16_t*)(ws + 2 * seg);
  bf16_t* aob = (bf16_t*)(ws + 3 * seg);
  bf16_t* WtQ = (bf16_t*)(ws + 4 * seg);
  bf16_t* WtK = (bf16_t*)(ws + 4 * seg + wseg);
  bf16_t* WtV = (bf16_t*)(ws + 4 * seg + 2 * wseg);
  bf16_t* WtO = (bf16_t*)(ws + 4 * seg + 3 * wseg);

  dim3 blk(256);
  dim3 wgrid(N / 32, K / 32);  // (32, 32)
  wconv_kernel<<<wgrid, blk, 0, stream>>>(Wq, WtQ, K, N);
  wconv_kernel<<<wgrid, blk, 0, stream>>>(Wk, WtK, K, N);
  wconv_kernel<<<wgrid, blk, 0, stream>>>(Wv, WtV, K, N);
  wconv_kernel<<<wgrid, blk, 0, stream>>>(Wo, WtO, K, N);

  dim3 ggrid(N / BN, M / BM);  // (8, 512)
  gemm_wmma2<0, 1, 0><<<ggrid, blk, 0, stream>>>(query, WtQ, bq, qbf, nullptr, M, N, K);
  gemm_wmma2<0, 1, 0><<<ggrid, blk, 0, stream>>>(key,   WtK, bk, kbf, nullptr, M, N, K);
  gemm_wmma2<0, 0, 0><<<ggrid, blk, 0, stream>>>(value, WtV, bv, vbf, nullptr, M, N, K);

  attn_head_kernel<<<dim3(16384), blk, 0, stream>>>(qbf, kbf, vbf, aob);

  gemm_wmma2<1, 0, 1><<<ggrid, blk, 0, stream>>>(aob, WtO, bo, nullptr, (float*)d_out, M, N, K);
}
